// GRU_38010460570431
// MI455X (gfx1250) — matrix operand order repaired and checked by reading, not compiled, not run on hardware
//
#include <hip/hip_runtime.h>
#include <hip/hip_bf16.h>
#include <math.h>

#define SEQ_LEN 16384
#define H       512
#define G3      1536   // 3*H
#define NWG     32     // persistent workgroups for the scan
#define RPW     48     // W_hh rows per workgroup (3 gates x 16 h-indices)

typedef __attribute__((ext_vector_type(16))) __bf16 v16bf;
typedef __attribute__((ext_vector_type(8)))  __bf16 v8bf;
typedef __attribute__((ext_vector_type(8)))  float  v8f;

static __device__ __forceinline__ v8f wmma_bf16(v16bf a, v16bf b, v8f c) {
  // 8 args: (neg_a, A, neg_b, B, c_mod, C, reuse_a, reuse_b)
  return __builtin_amdgcn_wmma_f32_16x16x32_bf16(false, a, false, b, (short)0, c,
                                                 false, false);
}

// ---------------- fp32 -> bf16 conversion ----------------
__global__ void cvt_f32_bf16(const float* __restrict__ in, __bf16* __restrict__ out, int n) {
  int i = blockIdx.x * blockDim.x + threadIdx.x;
  const int stride = gridDim.x * blockDim.x;
  for (; i < n; i += stride) out[i] = (__bf16)in[i];
}

// ---------------- zero h double-buffer + barrier counter ----------------
__global__ void init_state(float* h2, unsigned* cnt) {
  const int i = threadIdx.x;
  if (i == 0) *cnt = 0u;
  h2[i] = 0.f;
  h2[i + H] = 0.f;
}

// ---------------- gi = x @ W_ih^T + b_ih  (bf16 WMMA, f32 accum) ----------------
// grid = (G3/256, SEQ_LEN/64), block = 256 (8 waves); each wave owns a 32x64 tile
// (2 A frags reused across 4 B frags -> 8 WMMAs per K-step).
__global__ void __launch_bounds__(256)
gemm_gi(const __bf16* __restrict__ X, const __bf16* __restrict__ W,
        const float* __restrict__ bih, float* __restrict__ gi) {
  const int lane = threadIdx.x & 31;
  const int wave = threadIdx.x >> 5;
  const int wm = wave & 1;        // 2 waves along M (32 rows each)
  const int wn = wave >> 1;       // 4 waves along N (64 cols each)
  const int m0 = blockIdx.y * 64 + wm * 32;
  const int n0 = blockIdx.x * 256 + wn * 64;

  const int mrow0 = m0 + (lane & 15);
  const int koffA = (lane >> 4) * 8;      // A: K base 0 / 8 (and +16)
  const int koffB = (lane >> 4) * 8;      // B: K base 0 / 8 (and +16)

  const __bf16* pa0 = X + (size_t)mrow0 * H + koffA;
  const __bf16* pa1 = X + (size_t)(mrow0 + 16) * H + koffA;
  const __bf16* pb[4];
#pragma unroll
  for (int ci = 0; ci < 4; ++ci)
    pb[ci] = W + (size_t)(n0 + ci * 16 + (lane & 15)) * H + koffB;

  v8f c0[4] = {{}, {}, {}, {}};
  v8f c1[4] = {{}, {}, {}, {}};
  for (int kk = 0; kk < H; kk += 32) {
    v16bf a0, a1;
    *((v8bf*)&a0)       = *(const v8bf*)(pa0 + kk);
    *(((v8bf*)&a0) + 1) = *(const v8bf*)(pa0 + kk + 16);
    *((v8bf*)&a1)       = *(const v8bf*)(pa1 + kk);
    *(((v8bf*)&a1) + 1) = *(const v8bf*)(pa1 + kk + 16);
    v16bf b[4];
#pragma unroll
    for (int ci = 0; ci < 4; ++ci) {
      *((v8bf*)&b[ci])       = *(const v8bf*)(pb[ci] + kk);
      *(((v8bf*)&b[ci]) + 1) = *(const v8bf*)(pb[ci] + kk + 16);
    }
    if (kk + 32 < H) {
      __builtin_prefetch(pa0 + kk + 32, 0, 1);   // global_prefetch_b8
      __builtin_prefetch(pb[0] + kk + 32, 0, 1);
      __builtin_prefetch(pb[2] + kk + 32, 0, 1);
    }
#pragma unroll
    for (int ci = 0; ci < 4; ++ci) {
      c0[ci] = wmma_bf16(a0, b[ci], c0[ci]);
      c1[ci] = wmma_bf16(a1, b[ci], c1[ci]);
    }
  }

  // D layout: element v -> row = v + 8*(lane>=16), col = lane&15
  const int half = lane >> 4, ncol = lane & 15;
#pragma unroll
  for (int ci = 0; ci < 4; ++ci) {
    const float bias = bih[n0 + ci * 16 + ncol];
#pragma unroll
    for (int v = 0; v < 8; ++v) {
      const int r = m0 + half * 8 + v;
      gi[(size_t)r * G3 + n0 + ci * 16 + ncol]        = c0[ci][v] + bias;
      gi[(size_t)(r + 16) * G3 + n0 + ci * 16 + ncol] = c1[ci][v] + bias;
    }
  }
}

// ---------------- persistent GRU scan ----------------
// 32 workgroups x 256 threads; each WG owns h[wg*16 .. wg*16+15] and keeps its
// 48 W_hh rows (bf16) resident in LDS. 6 compute waves: 3 M-tiles x 2 K-halves
// -> per-step dependent WMMA chain is only 8 deep. One device barrier / step.
__global__ void __launch_bounds__(256)
gru_scan(const float* __restrict__ Whh, const float* __restrict__ bhh,
         const float* __restrict__ gi, float* __restrict__ h2,
         unsigned* __restrict__ cnt, float* __restrict__ out) {
  __shared__ __attribute__((aligned(32))) __bf16 Wl[RPW * H];  // 48 KB
  __shared__ __attribute__((aligned(32))) __bf16 hbf[H];
  __shared__ float hf[H];
  __shared__ float ghp[2][RPW];   // K-half partial sums
  __shared__ float bl[RPW];

  const int tid  = threadIdx.x;
  const int lane = tid & 31;
  const int wave = tid >> 5;
  const int wg   = blockIdx.x;
  const int idx0 = wg * 16;

  // Preload W slice: LDS row rr = g*16+j  <->  W_hh row g*H + idx0 + j
  for (int e = tid; e < RPW * H; e += 256) {
    const int rr = e >> 9;          // /H
    const int k  = e & (H - 1);
    const int g  = rr >> 4;
    const int j  = rr & 15;
    Wl[e] = (__bf16)Whh[(size_t)(g * H + idx0 + j) * H + k];
  }
  if (tid < RPW) {
    const int g = tid >> 4, j = tid & 15;
    bl[tid] = bhh[g * H + idx0 + j];
  }
  __syncthreads();

  const int koffA = (lane >> 4) * 8;
  const int koffB = (lane >> 4) * 8;
  const int mt    = wave >> 1;            // M-tile 0..2 (waves 0..5)
  const int kh    = wave & 1;             // K-half
  const int arow  = (mt * 16 + (lane & 15)) * H + koffA;
  const int kbase = kh * (H / 2);

  for (int t = 0; t < SEQ_LEN; ++t) {
    const float* hin  = h2 + (t & 1) * H;          // h_{t-1}
    float*       hout = h2 + ((t + 1) & 1) * H;    // h_t

    // gi[t] depends only on t: issue these loads early so their latency
    // overlaps the h broadcast + WMMA phase.
    float gir = 0.f, giz = 0.f, gin = 0.f;
    if (tid < 16) {
      const float* git = gi + (size_t)t * G3 + idx0 + tid;
      gir = git[0];
      giz = git[H];
      gin = git[2 * H];
    }

    // Broadcast h into LDS (f32 + bf16); volatile to bypass stale L0 lines.
    for (int i = tid; i < H; i += 256) {
      const float v = *(volatile const float*)(hin + i);
      hf[i]  = v;
      hbf[i] = (__bf16)v;
    }
    __syncthreads();

    // gh = W_slice @ h via WMMA (h replicated across the 16 B columns).
    if (wave < 6) {            // wave-uniform branch: EXEC stays all-ones
      v8f acc = {};
      for (int kk = kbase; kk < kbase + H / 2; kk += 32) {
        v16bf a, b;
        *((v8bf*)&a)       = *(const v8bf*)&Wl[arow + kk];
        *(((v8bf*)&a) + 1) = *(const v8bf*)&Wl[arow + kk + 16];
        *((v8bf*)&b)       = *(const v8bf*)&hbf[kk + koffB];
        *(((v8bf*)&b) + 1) = *(const v8bf*)&hbf[kk + koffB + 16];
        acc = wmma_bf16(a, b, acc);
      }
      if ((lane & 15) == 0) {   // column 0 lives in lanes 0 and 16
        const int rb = mt * 16 + (lane >> 4) * 8;
#pragma unroll
        for (int v = 0; v < 8; ++v) ghp[kh][rb + v] = acc[v];
      }
    }
    __syncthreads();

    // Gates + state update for this WG's 16 hidden indices.
    if (tid < 16) {
      const float ghr = ghp[0][tid]      + ghp[1][tid]      + bl[tid];
      const float ghz = ghp[0][16 + tid] + ghp[1][16 + tid] + bl[16 + tid];
      const float ghn = ghp[0][32 + tid] + ghp[1][32 + tid] + bl[32 + tid];
      const float r = 1.f / (1.f + __expf(-(gir + ghr)));
      const float z = 1.f / (1.f + __expf(-(giz + ghz)));
      const float n = tanhf(gin + r * ghn);
      const int idx = idx0 + tid;
      const float hnew = (1.f - z) * n + z * hf[idx];
      out[(size_t)t * H + idx] = hnew;
      hout[idx] = hnew;
    }

    // Device-wide barrier (monotonic counter; no reset races).
    __threadfence();
    __syncthreads();
    if (tid == 0) {
      atomicAdd(cnt, 1u);
      const unsigned target = (unsigned)(t + 1) * NWG;
      while (__hip_atomic_load(cnt, __ATOMIC_ACQUIRE, __HIP_MEMORY_SCOPE_AGENT) < target) {
        __builtin_amdgcn_s_sleep(1);
      }
    }
    __syncthreads();
  }
}

// ---------------- launch ----------------
extern "C" void kernel_launch(void* const* d_in, const int* in_sizes, int n_in,
                              void* d_out, int out_size, void* d_ws, size_t ws_size,
                              hipStream_t stream) {
  (void)in_sizes; (void)n_in; (void)out_size; (void)ws_size;
  const float* x   = (const float*)d_in[0];
  const float* Wih = (const float*)d_in[1];
  const float* Whh = (const float*)d_in[2];
  const float* bih = (const float*)d_in[3];
  const float* bhh = (const float*)d_in[4];
  float* out = (float*)d_out;

  char* ws = (char*)d_ws;
  size_t o = 0;
  auto take = [&](size_t bytes) -> char* {
    char* p = ws + o;
    o += (bytes + 255) & ~(size_t)255;
    return p;
  };
  __bf16*   xbf = (__bf16*)take((size_t)SEQ_LEN * H * sizeof(__bf16));   // 16 MB
  __bf16*   wbf = (__bf16*)take((size_t)G3 * H * sizeof(__bf16));        // 1.5 MB
  float*    gi  = (float*)take((size_t)SEQ_LEN * G3 * sizeof(float));    // 96 MB (L2-resident)
  float*    h2  = (float*)take((size_t)2 * H * sizeof(float));
  unsigned* cnt = (unsigned*)take(256);

  cvt_f32_bf16<<<2048, 256, 0, stream>>>(x,   xbf, SEQ_LEN * H);
  cvt_f32_bf16<<<1024, 256, 0, stream>>>(Wih, wbf, G3 * H);
  init_state<<<1, 512, 0, stream>>>(h2, cnt);
  gemm_gi<<<dim3(G3 / 256, SEQ_LEN / 64), 256, 0, stream>>>(xbf, wbf, bih, gi);
  gru_scan<<<NWG, 256, 0, stream>>>(Whh, bhh, gi, h2, cnt, out);
}
